// Scanner_87419764342914
// MI455X (gfx1250) — compile-verified, matrix operand order checked by reading, not run on hardware
//
#include <hip/hip_runtime.h>
#include <hip/hip_bf16.h>

#define BB 32
#define SS 4096
#define CC 512
#define NOUT 30

typedef __attribute__((ext_vector_type(16))) __bf16 v16bf;
typedef __attribute__((ext_vector_type(8)))  float  v8f;
typedef __attribute__((ext_vector_type(8)))  unsigned short us8;
typedef __attribute__((ext_vector_type(16))) unsigned short us16;
typedef __attribute__((ext_vector_type(4)))  unsigned int v4u;
typedef __attribute__((ext_vector_type(8)))  int v8i;
typedef __attribute__((ext_vector_type(4)))  int v4i;

#if defined(__gfx1250__) && __has_builtin(__builtin_amdgcn_tensor_load_to_lds)
#define USE_TDM 1
#else
#define USE_TDM 0
#endif

__device__ __forceinline__ unsigned short f2bf(float f) {
    unsigned int u = __float_as_uint(f);
    u += 0x7FFFu + ((u >> 16) & 1u);   // round-to-nearest-even
    return (unsigned short)(u >> 16);
}

// Build fragment: 16 bf16 per lane = two contiguous 8-element (16B) LDS chunks.
__device__ __forceinline__ v16bf mkfrag(const unsigned short* p) {
    us8 lo = *(const us8*)(p);
    us8 hi = *(const us8*)(p + 16);
    us16 c = __builtin_shufflevector(lo, hi, 0,1,2,3,4,5,6,7,8,9,10,11,12,13,14,15);
    return __builtin_bit_cast(v16bf, c);
}

// Merge 10 conv kernels into one width-10, 32-col (30 used) bf16 weight tensor.
// Output layout: wbf[(d*32 + n)*512 + c], d = tap offset (x[s-d]), n = out col.
__global__ __launch_bounds__(256) void prep_weights(
    const float* __restrict__ w1, const float* __restrict__ w2,
    const float* __restrict__ w3, const float* __restrict__ w4,
    const float* __restrict__ w5, const float* __restrict__ w6,
    const float* __restrict__ w7, const float* __restrict__ w8,
    const float* __restrict__ w9, const float* __restrict__ w10,
    unsigned short* __restrict__ wbf)
{
    const float* ws[10] = {w1,w2,w3,w4,w5,w6,w7,w8,w9,w10};
    int idx = blockIdx.x * 256 + threadIdx.x;      // [0, 10*32*512)
    if (idx >= 10 * 32 * 512) return;
    int d = idx / (32 * 512);
    int n = (idx / 512) & 31;
    int c = idx & 511;
    float v = 0.f;
    if (n < NOUT) {
        int k = n / 3 + 1;       // conv width 1..10
        int p = n % 3;
        int t = (k - 1) - d;     // tap index inside w_k
        if (t >= 0) v = ws[k - 1][((size_t)t * CC + c) * 3 + p];
    }
    wbf[idx] = f2bf(v);
}

// One WG per (batch, 128 seq positions). 8 waves; wave w owns rows s0+16w..+15.
__global__ __launch_bounds__(256) void conv_wmma(
    const float* __restrict__ x, const unsigned short* __restrict__ wbf,
    float* __restrict__ out)
{
    __shared__ unsigned short xs[137 * 64];        // x tile (halo 9) bf16
    __shared__ unsigned short wsm[10 * 32 * 64];   // weight slab bf16

    const int tid  = threadIdx.x;
    const int lane = tid & 31;
    const int wave = tid >> 5;
    const int b    = blockIdx.y;
    const int s0   = blockIdx.x * 128;
    const int lm   = lane & 15;                 // row (A) / col (B) within 16
    const int g0   = (lane >> 4) * 8;           // K sub-offset per wave32 layout

    v8f acc0 = {};   // columns 0..15
    v8f acc1 = {};   // columns 16..31

    for (int cc = 0; cc < 8; ++cc) {            // 8 chunks of 64 channels
        __syncthreads();
        // stage x[s0-9 .. s0+127, cc*64 .. +64) -> bf16 LDS (needs f32->bf16)
        for (int idx = tid; idx < 137 * 64; idx += 256) {
            int r = idx >> 6, c = idx & 63;
            int gs = s0 - 9 + r;
            float v = (gs >= 0) ? x[((size_t)b * SS + gs) * CC + cc * 64 + c] : 0.f;
            xs[idx] = f2bf(v);
        }
#if USE_TDM
        // Tensor Data Mover: DMA W[0:320)[cc*64 .. +64) (bf16, 2D strided tile)
        // straight into LDS. One wave issues; TENSORcnt tracks completion.
        if (wave == 0) {
            unsigned long long ga = (unsigned long long)(uintptr_t)wbf
                                  + (unsigned long long)cc * 128ull; // cc*64 elem *2B
            unsigned lds_addr = (unsigned)(uintptr_t)&wsm[0];
            // D# group0: count=1, lds_addr, 57b global_addr, type=2 (bits 127:126)
            v4u g0 = { 1u,
                       lds_addr,
                       (unsigned)ga,
                       (unsigned)((ga >> 32) & 0x01FFFFFFull) | (2u << 30) };
            // D# group1: data_size=1(2B); tensor_dim0=512, tensor_dim1=320,
            // tile_dim0=64, tile_dim1=320, tensor_dim0_stride=512
            v8i g1 = { (int)(1u << 16),
                       (int)((512u & 0xFFFFu) << 16),
                       (int)(((512u >> 16) & 0xFFFFu) | ((320u & 0xFFFFu) << 16)),
                       (int)(((320u >> 16) & 0xFFFFu) | (64u << 16)),
                       (int)320,
                       (int)512,
                       0, 0 };
            v4i gz4 = { 0, 0, 0, 0 };
            v8i gz8 = { 0, 0, 0, 0, 0, 0, 0, 0 };
            __builtin_amdgcn_tensor_load_to_lds(g0, g1, gz4, gz4, gz8, 0);
#if __has_builtin(__builtin_amdgcn_s_wait_tensorcnt)
            __builtin_amdgcn_s_wait_tensorcnt(0);
#endif
        }
#else
        for (int idx = tid; idx < 10 * 32 * 64; idx += 256) {
            int dn = idx >> 6;
            int c  = idx & 63;
            wsm[idx] = wbf[(size_t)dn * 512 + cc * 64 + c];
        }
#endif
        __syncthreads();

        for (int t = 0; t < 10; ++t) {
            const unsigned short* xrow = &xs[(9 + wave * 16 + lm - t) * 64];
            const unsigned short* w0   = &wsm[(t * 32 + lm) * 64];
            const unsigned short* w1   = &wsm[(t * 32 + 16 + lm) * 64];
#pragma unroll
            for (int kc = 0; kc < 2; ++kc) {
                int off = kc * 32 + g0;
                v16bf a  = mkfrag(xrow + off);
                v16bf b0 = mkfrag(w0 + off);
                v16bf b1 = mkfrag(w1 + off);
                acc0 = __builtin_amdgcn_wmma_f32_16x16x32_bf16(
                    false, a, false, b0, (short)0, acc0, false, false);
                acc1 = __builtin_amdgcn_wmma_f32_16x16x32_bf16(
                    false, a, false, b1, (short)0, acc1, false, false);
            }
        }
    }

    // D layout: lanes 0-15: VGPR r -> M=r, N=lane; lanes 16-31: M=r+8, N=lane-16
    const int mofs  = (lane >> 4) * 8;
    const int sbase = s0 + wave * 16;
#pragma unroll
    for (int r = 0; r < 8; ++r) {
        int s = sbase + r + mofs;
        size_t base = ((size_t)b * SS + s) * NOUT;
        out[base + lm] = fmaxf(acc0[r], 0.f);
        int n1 = 16 + lm;
        if (n1 < NOUT) out[base + n1] = fmaxf(acc1[r], 0.f);
    }
}

extern "C" void kernel_launch(void* const* d_in, const int* in_sizes, int n_in,
                              void* d_out, int out_size, void* d_ws, size_t ws_size,
                              hipStream_t stream) {
    (void)in_sizes; (void)n_in; (void)out_size; (void)ws_size;
    const float* x = (const float*)d_in[0];
    unsigned short* wbf = (unsigned short*)d_ws;   // 10*32*512 bf16 = 320 KiB

    prep_weights<<<(10 * 32 * 512 + 255) / 256, 256, 0, stream>>>(
        (const float*)d_in[1], (const float*)d_in[2], (const float*)d_in[3],
        (const float*)d_in[4], (const float*)d_in[5], (const float*)d_in[6],
        (const float*)d_in[7], (const float*)d_in[8], (const float*)d_in[9],
        (const float*)d_in[10], wbf);

    dim3 grid(SS / 128, BB);
    conv_wmma<<<grid, 256, 0, stream>>>(x, wbf, (float*)d_out);
}
